// MeshConvNet_76141180223546
// MI455X (gfx1250) — matrix operand order repaired and checked by reading, not compiled, not run on hardware
//
#include <hip/hip_runtime.h>
#include <hip/hip_bf16.h>

// ---------------------------------------------------------------------------
// MeshConvNet on MI455X (gfx1250): 4 mesh-convs as bf16 WMMA GEMMs
//   M = 256 (out channels), N = B*E = 65536, K = C*7 (896 / 1792)
// ---------------------------------------------------------------------------

#define B_    4
#define E_    16384
#define N_    (B_ * E_)      // 65536
#define C0_   128
#define C1_   256
#define OC_   256
#define KJ_   7
#define KK0_  (KJ_ * C0_)    // 896  = 28 * 32
#define KK1_  (KJ_ * C1_)    // 1792 = 56 * 32
#define NEG_SLOPE 0.01f
#define BN_EPS    1e-5f

typedef __bf16 bf16;
typedef __attribute__((ext_vector_type(16))) __bf16 v16bf;
typedef __attribute__((ext_vector_type(8)))  __bf16 v8bf;
typedef __attribute__((ext_vector_type(8)))  float  v8f;

__device__ __forceinline__ float leaky(float v) {
    return v > 0.0f ? v : NEG_SLOPE * v;
}

// ---------------------------------------------------------------------------
// Repack weights W[o][c][j] (f32) -> Wt bf16 tiles [kc][256][32],
// contraction index kk = j*C + c  (A-operand side of the GEMM).
// ---------------------------------------------------------------------------
__global__ void pack_weights(const float* __restrict__ W, bf16* __restrict__ Wt, int C) {
    const int KK  = KJ_ * C;
    const int idx = blockIdx.x * blockDim.x + threadIdx.x;
    if (idx >= OC_ * KK) return;
    const int o  = idx / KK;
    const int kk = idx - o * KK;
    const int j  = kk / C;
    const int c  = kk - j * C;
    const float v = W[((size_t)o * C + c) * KJ_ + j];
    const int kc = kk >> 5, ko = kk & 31;
    Wt[(((size_t)kc * OC_ + o) << 5) + ko] = (bf16)v;
}

// ---------------------------------------------------------------------------
// Channel transpose with fused leaky+BN:
//   src[b*sb + c*sc + e]  ->  dst[(b*E+e)*C + c]
// Makes every gather tap in build_G a contiguous 128B wave read.
// ---------------------------------------------------------------------------
__global__ void transpose_bn(const float* __restrict__ src,
                             float* __restrict__ dst,
                             const float* __restrict__ stats,  // [0..255]=mean,[256..]=rstd
                             const float* __restrict__ gamma,
                             const float* __restrict__ beta,
                             int C, size_t sb, size_t sc, int use_bn) {
    __shared__ float tile[32][33];
    const int b  = blockIdx.z;
    const int c0 = blockIdx.y << 5;
    const int e0 = blockIdx.x << 5;
    const int tx = threadIdx.x & 31;
    const int ty = threadIdx.x >> 5;          // 0..7
    const float* s = src + (size_t)b * sb;
#pragma unroll
    for (int i = 0; i < 32; i += 8) {
        const int c = c0 + ty + i;
        float v = s[(size_t)c * sc + e0 + tx];
        if (use_bn) {
            v = leaky(v);
            const float scale = stats[256 + c] * gamma[c];
            v = (v - stats[c]) * scale + beta[c];
        }
        tile[ty + i][tx] = v;
    }
    __syncthreads();
#pragma unroll
    for (int i = 0; i < 32; i += 8)
        dst[((size_t)b * E_ + e0 + ty + i) * C + c0 + tx] = tile[tx][ty + i];
}

// ---------------------------------------------------------------------------
// Build the G (B-operand) matrix in bf16, fragment-ready tiled layout:
//   Gt[kc][n][32], kk = j*C + c, n = b*E + e.
// One wave per edge n; lanes = 32 consecutive channels; all 5 taps are
// contiguous 128B reads from the channel-transposed activations xT[n][c].
// ---------------------------------------------------------------------------
__global__ void build_G(const float* __restrict__ xT,     // [N][C]
                        const int*   __restrict__ edges,  // [B][E][4]
                        bf16* __restrict__ Gt,
                        int C) {
    const int wid  = (blockIdx.x * blockDim.x + threadIdx.x) >> 5;   // = n
    const int lane = threadIdx.x & 31;
    if (wid >= N_) return;
    const int n = wid;
    const int b = n >> 14;           // E_ = 16384
    const int e = n & (E_ - 1);

    const int4 eg = *(const int4*)(edges + ((size_t)b * E_ + e) * 4);
    const size_t bofs = (size_t)b * E_;
    const float* r0 = xT + (size_t)n * C;
    const float* r1 = xT + (bofs + eg.x) * C;
    const float* r2 = xT + (bofs + eg.y) * C;
    const float* r3 = xT + (bofs + eg.z) * C;
    const float* r4 = xT + (bofs + eg.w) * C;
    const int cchunks = C >> 5;

    for (int cc = 0; cc < cchunks; ++cc) {
        const int c = (cc << 5) + lane;
        const float f0 = r0[c];
        const float f1 = r1[c];
        const float f2 = r2[c];
        const float f3 = r3[c];
        const float f4 = r4[c];

        const float s13 = f1 + f3, s24 = f2 + f4;
        const float g5 = s13 + s24;
        const float avg = 0.25f * g5;
        const float d1 = f1 - avg, d2 = f2 - avg, d3 = f3 - avg, d4 = f4 - avg;
        const float g[KJ_] = {
            f0, s13, s24, fabsf(f1 - f3), fabsf(f2 - f4), g5,
            d1 * d1 + d2 * d2 + d3 * d3 + d4 * d4
        };
#pragma unroll
        for (int j = 0; j < KJ_; ++j) {
            const int kc = j * cchunks + cc;      // == (j*C + c) >> 5
            Gt[(((size_t)kc * N_ + n) << 5) + lane] = (bf16)g[j];
        }
    }
}

// ---------------------------------------------------------------------------
// Per-channel BatchNorm stats of y = leaky(h):  one block per channel m.
//   stats[m] = mean, stats[256+m] = rsqrt(var + eps)
// ---------------------------------------------------------------------------
__global__ void bn_stats(const float* __restrict__ h, float* __restrict__ stats) {
    const int m = blockIdx.x;
    const int t = threadIdx.x;                       // 256 threads
    const float4* row = (const float4*)(h + (size_t)m * N_);
    float s = 0.0f, ss = 0.0f;
    for (int i = t; i < N_ / 4; i += 256) {
        const float4 v = row[i];
        float y;
        y = leaky(v.x); s += y; ss += y * y;
        y = leaky(v.y); s += y; ss += y * y;
        y = leaky(v.z); s += y; ss += y * y;
        y = leaky(v.w); s += y; ss += y * y;
    }
#pragma unroll
    for (int o = 16; o > 0; o >>= 1) {               // wave32 reduction
        s  += __shfl_down(s,  o, 32);
        ss += __shfl_down(ss, o, 32);
    }
    __shared__ float red[16];
    const int w = t >> 5, lane = t & 31;
    if (lane == 0) { red[w] = s; red[8 + w] = ss; }
    __syncthreads();
    if (t == 0) {
        s = 0.0f; ss = 0.0f;
#pragma unroll
        for (int i = 0; i < 8; ++i) { s += red[i]; ss += red[8 + i]; }
        const float mean = s * (1.0f / N_);
        const float var  = ss * (1.0f / N_) - mean * mean;
        stats[m]       = mean;
        stats[256 + m] = rsqrtf(var + BN_EPS);
    }
}

// ---------------------------------------------------------------------------
// bf16 WMMA GEMM, register double-buffered over k-chunks, single base
// pointer per operand (fragment rows reached via 24-bit instruction ioffsets).
//   256 threads = 8 waves (2 x 4); each wave: 64x64 tile = 16 accumulators.
// Epilogue modes:
//   0: h[m][n] = acc
//   1: h[m][n] = acc and h1[m][n] = acc            (conv0: save residual)
//   2: out[b][o][e] = leaky(acc + h1[m][n])        (last conv: fused output)
// ---------------------------------------------------------------------------
__global__ void __launch_bounds__(256)
wmma_gemm(const bf16* __restrict__ Wt,   // [kchunks][256][32]
          const bf16* __restrict__ Gt,   // [kchunks][N_][32]
          float* __restrict__ h,         // [256][N_]
          const float* __restrict__ h1,  // residual (mode 2)
          float* __restrict__ h1w,       // residual store (mode 1)
          float* __restrict__ out,       // [B][256][E] (mode 2)
          int kchunks, int mode) {       // kchunks always even (28 or 56)
    const int lane = threadIdx.x & 31;
    const int w    = threadIdx.x >> 5;      // 0..7
    const int mw   = w & 1;                 // 2 wave-rows (M)
    const int nw   = w >> 1;                // 4 wave-cols (N)
    const int mBase = blockIdx.y * 128 + mw * 64;
    const int nBase = blockIdx.x * 256 + nw * 64;
    const int lm = lane & 15;
    const int lh = lane >> 4;

    const size_t aStep = (size_t)OC_ << 5;  // elements per k-chunk of Wt
    const size_t bStep = (size_t)N_ << 5;   // elements per k-chunk of Gt

    // Single base pointer per operand; fragment rows are +512*i elements
    // (= 1024*i bytes) -> folded into the load's immediate offset.
    const bf16* aBase = Wt + ((size_t)(mBase + lm) << 5) + 16 * lh; // A 32B run
    const bf16* bBase = Gt + ((size_t)(nBase + lm) << 5) + 16 * lh; // B 32B run

    v8f acc[4][4] = {};

    auto loadA = [&](const bf16* p, v16bf (&a)[4]) {
#pragma unroll
        for (int i = 0; i < 4; ++i) {
            a[i] = *(const v16bf*)(p + 512 * i);                // K[16lh..16lh+15]
        }
    };
    auto loadB = [&](const bf16* p, v16bf (&bm)[4]) {
#pragma unroll
        for (int j = 0; j < 4; ++j) {
            bm[j] = *(const v16bf*)(p + 512 * j);
            __builtin_prefetch(p + 512 * j + 2 * bStep, 0, 1);  // speculative
        }
    };
    auto domm = [&](v16bf (&a)[4], v16bf (&bm)[4]) {
#pragma unroll
        for (int i = 0; i < 4; ++i)
#pragma unroll
            for (int j = 0; j < 4; ++j)
                acc[i][j] = __builtin_amdgcn_wmma_f32_16x16x32_bf16(
                    false, a[i], false, bm[j], (short)0, acc[i][j], false, false);
    };

    v16bf a0[4], b0[4], a1[4], b1[4];
    loadA(aBase, a0); loadB(bBase, b0);
    for (int kc = 0; kc + 2 <= kchunks; kc += 2) {
        loadA(aBase + (kc + 1) * aStep, a1);      // fill other buffer
        loadB(bBase + (kc + 1) * bStep, b1);
        domm(a0, b0);                              // waits only on old loads
        if (kc + 2 < kchunks) {
            loadA(aBase + (kc + 2) * aStep, a0);
            loadB(bBase + (kc + 2) * bStep, b0);
        }
        domm(a1, b1);
    }

    // C/D layout: VGPR r -> M = r (lanes 0-15) / r+8 (lanes 16-31), N = lane%16
    if (mode != 2) {
#pragma unroll
        for (int i = 0; i < 4; ++i)
#pragma unroll
            for (int j = 0; j < 4; ++j)
#pragma unroll
                for (int r = 0; r < 8; ++r) {
                    const int m = mBase + i * 16 + r + 8 * lh;
                    const int n = nBase + j * 16 + lm;
                    h[(size_t)m * N_ + n] = acc[i][j][r];
                    if (mode == 1) h1w[(size_t)m * N_ + n] = acc[i][j][r];
                }
    } else {
#pragma unroll
        for (int i = 0; i < 4; ++i)
#pragma unroll
            for (int j = 0; j < 4; ++j)
#pragma unroll
                for (int r = 0; r < 8; ++r) {
                    const int m = mBase + i * 16 + r + 8 * lh;
                    const int n = nBase + j * 16 + lm;
                    const int b = n >> 14;            // E_ = 16384
                    const int e = n & (E_ - 1);
                    const float v = leaky(acc[i][j][r] + h1[(size_t)m * N_ + n]);
                    out[((size_t)b * OC_ + m) * E_ + e] = v;
                }
    }
}

// ---------------------------------------------------------------------------
extern "C" void kernel_launch(void* const* d_in, const int* in_sizes, int n_in,
                              void* d_out, int out_size, void* d_ws, size_t ws_size,
                              hipStream_t stream) {
    const float* x      = (const float*)d_in[0];   // [B, 128, E, 1]
    const int*   edges  = (const int*)d_in[1];     // [B, E, 4]
    const float* W0     = (const float*)d_in[2];   // [256, 128, 7]
    const float* Ws     = (const float*)d_in[3];   // [3, 256, 256, 7]
    const float* gammas = (const float*)d_in[4];   // [3, 256]
    const float* betas  = (const float*)d_in[5];   // [3, 256]
    float* out = (float*)d_out;

    char* ws = (char*)d_ws;
    bf16* Gt = (bf16*)ws;                               // 224 MiB
    size_t off = (size_t)KK1_ * N_ * sizeof(bf16);
    float* h   = (float*)(ws + off); off += (size_t)OC_ * N_ * sizeof(float);  // 64 MiB
    float* h1  = (float*)(ws + off); off += (size_t)OC_ * N_ * sizeof(float);  // 64 MiB
    float* xT  = (float*)(ws + off); off += (size_t)OC_ * N_ * sizeof(float);  // 64 MiB
    bf16*  Wt  = (bf16*)(ws + off);  off += (size_t)KK1_ * OC_ * sizeof(bf16);
    float* stats = (float*)(ws + off);

    const dim3 ggrid(N_ / 256, OC_ / 128);

    // ---- conv0: 128 -> 256; epilogue also saves the residual h1 -------------
    transpose_bn<<<dim3(E_ / 32, C0_ / 32, B_), 256, 0, stream>>>(
        x, xT, nullptr, nullptr, nullptr, C0_, (size_t)C0_ * E_, (size_t)E_, 0);
    pack_weights<<<(OC_ * KK0_ + 255) / 256, 256, 0, stream>>>(W0, Wt, C0_);
    build_G<<<N_ / 8, 256, 0, stream>>>(xT, edges, Gt, C0_);
    wmma_gemm<<<ggrid, 256, 0, stream>>>(Wt, Gt, h, nullptr, h1, nullptr,
                                         KK0_ / 32, 1);

    // ---- 3 skip layers: leaky+BN fused into the channel transpose;
    //      last GEMM fuses residual + final leaky + output layout ------------
    for (int i = 0; i < 3; ++i) {
        bn_stats<<<OC_, 256, 0, stream>>>(h, stats);
        // h layout: [c][b*E + e]  ->  sb = E_, sc = N_
        transpose_bn<<<dim3(E_ / 32, C1_ / 32, B_), 256, 0, stream>>>(
            h, xT, stats, gammas + i * OC_, betas + i * OC_,
            C1_, (size_t)E_, (size_t)N_, 1);
        pack_weights<<<(OC_ * KK1_ + 255) / 256, 256, 0, stream>>>(
            Ws + (size_t)i * OC_ * C1_ * KJ_, Wt, C1_);
        build_G<<<N_ / 8, 256, 0, stream>>>(xT, edges, Gt, C1_);
        const int mode = (i == 2) ? 2 : 0;
        wmma_gemm<<<ggrid, 256, 0, stream>>>(Wt, Gt, h, h1, nullptr, out,
                                             KK1_ / 32, mode);
    }
}
